// RelativisticAttention_78013785964855
// MI455X (gfx1250) — matrix operand order repaired and checked by reading, not compiled, not run on hardware
//
#include <hip/hip_runtime.h>
#include <hip/hip_bf16.h>

typedef __attribute__((ext_vector_type(16))) _Float16 v16h;
typedef __attribute__((ext_vector_type(8)))  _Float16 v8h;
typedef __attribute__((ext_vector_type(8)))  float    v8f;

#define S_   2048
#define HID_ 2048
#define H_   16
#define D_   128
#define NE_  (HID_*HID_)      // 4194304 elements per matrix

#define WMMA_F16(a,b,c) __builtin_amdgcn_wmma_f32_16x16x32_f16(false,(a),false,(b),(short)0,(c),false,false)

static __device__ __forceinline__ v16h pack16(v8h lo, v8h hi) {
  return __builtin_shufflevector(lo, hi, 0,1,2,3,4,5,6,7,8,9,10,11,12,13,14,15);
}

// Stage a 128-row x 32-col f16 weight tile into LDS via the async-to-LDS path.
// 128 threads x 4 chunks x 32 lanes-worth: each lane copies 16B; ASYNCcnt tracks it.
static __device__ __forceinline__ void stage_tile(const _Float16* __restrict__ Wbase,
                                                  int kk, unsigned ldsoff, int tid) {
#pragma unroll
  for (int i = 0; i < 4; ++i) {
    const int c = i * 128 + tid;                       // chunk index: 4 x 16B per row
    const unsigned loff = ldsoff + (unsigned)(c * 16); // LDS byte offset, row-major [128][32]
    const _Float16* g = Wbase + (size_t)(c >> 2) * HID_ + kk + (c & 3) * 8;
    asm volatile("global_load_async_to_lds_b128 %0, %1, off"
                 :: "v"(loff), "v"((unsigned long long)(uintptr_t)g) : "memory");
  }
}

// ---------------------------------------------------------------- prep: fp32->f16 + speed
__global__ void prep_kernel(const float* __restrict__ x,  const float* __restrict__ imp,
                            const float* __restrict__ wq, const float* __restrict__ wk,
                            const float* __restrict__ wv, const float* __restrict__ wo,
                            _Float16* __restrict__ xh,  _Float16* __restrict__ wqh,
                            _Float16* __restrict__ wkh, _Float16* __restrict__ wvh,
                            _Float16* __restrict__ woh, float* __restrict__ speed) {
  int i = blockIdx.x * blockDim.x + threadIdx.x;
  if (i < NE_) {
    xh[i]  = (_Float16)x[i];
    wqh[i] = (_Float16)wq[i];
    wkh[i] = (_Float16)wk[i];
    wvh[i] = (_Float16)wv[i];
    woh[i] = (_Float16)wo[i];
  }
  if (i < S_) {
    float t = (imp[i] - 0.5f) * 10.0f;
    speed[i] = 0.5f + 1.5f / (1.0f + __expf(-t));
  }
}

// ---------------- QKV GEMM + RoPE: block = 128 rows x 1 head, weight tile shared via LDS
__global__ __launch_bounds__(128, 1)
void qkv_rope_kernel(const _Float16* __restrict__ xh,
                     const _Float16* __restrict__ wqh, const _Float16* __restrict__ wkh,
                     const _Float16* __restrict__ wvh,
                     const float* __restrict__ cosT, const float* __restrict__ sinT,
                     _Float16* __restrict__ Qh, _Float16* __restrict__ Kh,
                     _Float16* __restrict__ Vt) {
  __shared__ _Float16 wtile[2][128 * 32];   // double-buffered 8KB weight tiles
  const int tid = threadIdx.x;
  const int w = tid >> 5, lane = tid & 31;
  const int li = lane & 15, half = lane >> 4;

  const int BPM = (S_ / 128) * H_;          // 256 blocks per matrix
  const int mat = blockIdx.x / BPM;         // 0=Q 1=K 2=V
  const int rem = blockIdx.x % BPM;
  const int rb  = rem / H_, h = rem % H_;
  const int s0  = rb * 128 + w * 32;        // this wave's 32-row strip

  const _Float16* __restrict__ W = (mat == 0) ? wqh : (mat == 1 ? wkh : wvh);
  const _Float16* __restrict__ Wbase = W + (size_t)h * D_ * HID_;
  const _Float16* __restrict__ aBase0 = xh + (size_t)(s0 + li) * HID_;
  const _Float16* __restrict__ aBase1 = xh + (size_t)(s0 + 16 + li) * HID_;

  const unsigned lds0 = (unsigned)(uintptr_t)&wtile[0][0];
  const unsigned lds1 = (unsigned)(uintptr_t)&wtile[1][0];

  v8f acc[2][8] = {};
  stage_tile(Wbase, 0, lds0, tid);
  for (int kk = 0; kk < HID_; kk += 32) {
    const int buf = (kk >> 5) & 1;
    if (kk + 32 < HID_) {
      stage_tile(Wbase, kk + 32, buf ? lds0 : lds1, tid);   // prefetch next tile
      asm volatile("s_wait_asynccnt 4" ::: "memory");       // current tile landed
    } else {
      asm volatile("s_wait_asynccnt 0" ::: "memory");
    }
    __syncthreads();                                        // all waves' stages visible

    const int ka = kk + half * 8;
    v16h a0 = pack16(*(const v8h*)(aBase0 + ka), *(const v8h*)(aBase0 + ka + 16));
    v16h a1 = pack16(*(const v8h*)(aBase1 + ka), *(const v8h*)(aBase1 + ka + 16));
    const _Float16* Bl = &wtile[buf][0];
#pragma unroll
    for (int t = 0; t < 8; ++t) {
      const _Float16* bp = Bl + (t * 16 + li) * 32 + half * 8;
      v16h b = pack16(*(const v8h*)bp, *(const v8h*)(bp + 16));
      acc[0][t] = WMMA_F16(a0, b, acc[0][t]);   // B fragment shared across row groups
      acc[1][t] = WMMA_F16(a1, b, acc[1][t]);
    }
    __syncthreads();                            // done reading before tile is overwritten
  }

  if (mat < 2) {
    // RoPE in-register: dims d and d+64 are accumulator tiles t and t+4, same lane.
#pragma unroll
    for (int r = 0; r < 2; ++r)
#pragma unroll
      for (int t = 0; t < 4; ++t)
#pragma unroll
        for (int j = 0; j < 8; ++j) {
          const int srow = s0 + r * 16 + j + half * 8;
          const int d    = t * 16 + li;
          const float c1 = cosT[srow * D_ + d],      sn1 = sinT[srow * D_ + d];
          const float c2 = cosT[srow * D_ + d + 64], sn2 = sinT[srow * D_ + d + 64];
          const float q1 = acc[r][t][j], q2 = acc[r][t + 4][j];
          acc[r][t][j]     = q1 * c1 - q2 * sn1;   // rotate_half: first half gets -x[d+64]
          acc[r][t + 4][j] = q2 * c2 + q1 * sn2;   // second half gets +x[d]
        }
    _Float16* __restrict__ dst = (mat == 0) ? Qh : Kh;   // layout (H,S,D)
#pragma unroll
    for (int r = 0; r < 2; ++r)
#pragma unroll
      for (int t = 0; t < 8; ++t)
#pragma unroll
        for (int j = 0; j < 8; ++j)
          dst[(size_t)(h * S_ + s0 + r * 16 + j + half * 8) * D_ + t * 16 + li] =
              (_Float16)acc[r][t][j];
  } else {
    // V stored transposed: (H, D, S) so flash PV B-fragments are contiguous.
#pragma unroll
    for (int r = 0; r < 2; ++r)
#pragma unroll
      for (int t = 0; t < 8; ++t)
#pragma unroll
        for (int j = 0; j < 8; ++j)
          Vt[(size_t)(h * D_ + t * 16 + li) * S_ + s0 + r * 16 + j + half * 8] =
              (_Float16)acc[r][t][j];
  }
}

// ------------------------------------------------- flash attention, one wave = 16 queries x 1 head
__global__ __launch_bounds__(128, 1)
void flash_kernel(const _Float16* __restrict__ Qh, const _Float16* __restrict__ Kh,
                  const _Float16* __restrict__ Vt, const float* __restrict__ speed,
                  _Float16* __restrict__ attn) {
  __shared__ _Float16 plds[4][16 * 32];
  const int wv = threadIdx.x >> 5;
  const int lane = threadIdx.x & 31;
  const int li = lane & 15, half = lane >> 4;
  const int wave = blockIdx.x * 4 + wv;
  const int qt = wave / H_, h = wave % H_;
  const int s0 = qt * 16;

  const _Float16* __restrict__ Qb = Qh + (size_t)(h * S_ + s0) * D_;
  const _Float16* __restrict__ Kb = Kh + (size_t)h * S_ * D_;
  const _Float16* __restrict__ Vb = Vt + (size_t)h * D_ * S_;

  // Q A-fragments, loop-invariant (d split into 4 K-steps of 32)
  v16h qa[4];
#pragma unroll
  for (int dd = 0; dd < 4; ++dd) {
    const _Float16* p = Qb + li * D_ + dd * 32 + half * 16;
    qa[dd] = pack16(*(const v8h*)p, *(const v8h*)(p + 8));
  }

  float m[8], l[8], fac[8];
#pragma unroll
  for (int j = 0; j < 8; ++j) {
    m[j] = -1e30f; l[j] = 0.0f;
    fac[j] = speed[s0 + j + half * 8] * 0.08838834764831845f;  // speed[q]/sqrt(128)
  }
  v8f o[8] = {};

  for (int kb = 0; kb < s0 + 16; kb += 32) {
    // ---- scores: 16 q x 32 k
    v8f sc[2] = {};
#pragma unroll
    for (int dd = 0; dd < 4; ++dd) {
      const int dw = dd * 32 + half * 16;
#pragma unroll
      for (int u = 0; u < 2; ++u) {
        v16h b = *(const v16h*)(Kb + (size_t)(kb + u * 16 + li) * D_ + dw);
        sc[u] = WMMA_F16(qa[dd], b, sc[u]);
      }
    }
    __builtin_prefetch(Kb + (size_t)(kb + 32 + li) * D_, 0, 1);

    // ---- scale + causal mask
#pragma unroll
    for (int u = 0; u < 2; ++u)
#pragma unroll
      for (int j = 0; j < 8; ++j) {
        const int kg = kb + u * 16 + li;
        const int qg = s0 + j + half * 8;
        float v = sc[u][j] * fac[j];
        sc[u][j] = (kg > qg) ? -1e30f : v;
      }

    // ---- online softmax (row reductions across 16 lanes of each half-wave)
    float sf[8];
#pragma unroll
    for (int j = 0; j < 8; ++j) {
      float mx = fmaxf(sc[0][j], sc[1][j]);
      mx = fmaxf(mx, __shfl_xor(mx, 8, 16));
      mx = fmaxf(mx, __shfl_xor(mx, 4, 16));
      mx = fmaxf(mx, __shfl_xor(mx, 2, 16));
      mx = fmaxf(mx, __shfl_xor(mx, 1, 16));
      const float mn = fmaxf(m[j], mx);
      sf[j] = __expf(m[j] - mn);
      m[j] = mn;
    }
#pragma unroll
    for (int u = 0; u < 2; ++u)
#pragma unroll
      for (int j = 0; j < 8; ++j) sc[u][j] = __expf(sc[u][j] - m[j]);
#pragma unroll
    for (int j = 0; j < 8; ++j) {
      float sm = sc[0][j] + sc[1][j];
      sm += __shfl_xor(sm, 8, 16);
      sm += __shfl_xor(sm, 4, 16);
      sm += __shfl_xor(sm, 2, 16);
      sm += __shfl_xor(sm, 1, 16);
      l[j] = l[j] * sf[j] + sm;
    }
#pragma unroll
    for (int t = 0; t < 8; ++t)
#pragma unroll
      for (int j = 0; j < 8; ++j) o[t][j] *= sf[j];

    // ---- C-tile -> A-fragment relayout of P through LDS (wave-private 1KB)
    _Float16* P = plds[wv];
#pragma unroll
    for (int u = 0; u < 2; ++u)
#pragma unroll
      for (int j = 0; j < 8; ++j)
        P[(j + half * 8) * 32 + u * 16 + li] = (_Float16)sc[u][j];
    asm volatile("s_wait_dscnt 0" ::: "memory");
    const _Float16* pr = P + li * 32 + half * 16;
    v16h pa = pack16(*(const v8h*)pr, *(const v8h*)(pr + 8));

    // ---- O += P @ V  (V transposed: contiguous B fragments)
#pragma unroll
    for (int t = 0; t < 8; ++t) {
      v16h b = *(const v16h*)(Vb + (size_t)(t * 16 + li) * S_ + kb + half * 16);
      o[t] = WMMA_F16(pa, b, o[t]);
    }
    asm volatile("s_wait_dscnt 0" ::: "memory");  // LDS reads done before next-iter stores
  }

  // ---- normalize + store as f16 (S, HID) for output projection
#pragma unroll
  for (int t = 0; t < 8; ++t)
#pragma unroll
    for (int j = 0; j < 8; ++j)
      attn[(size_t)(s0 + j + half * 8) * HID_ + h * D_ + t * 16 + li] =
          (_Float16)(o[t][j] / l[j]);
}

// -------------- output projection: out = attn @ Wo^T, block = 128 rows x 128 cols, LDS-staged
__global__ __launch_bounds__(128, 1)
void out_proj_kernel(const _Float16* __restrict__ attn, const _Float16* __restrict__ woh,
                     float* __restrict__ out) {
  __shared__ _Float16 wtile[2][128 * 32];
  const int tid = threadIdx.x;
  const int w = tid >> 5, lane = tid & 31;
  const int li = lane & 15, half = lane >> 4;
  const int rb = blockIdx.x / 16, g = blockIdx.x % 16;   // 128-row block x 128-col group
  const int s0 = rb * 128 + w * 32;

  const _Float16* __restrict__ Wbase = woh + (size_t)g * D_ * HID_;
  const _Float16* __restrict__ aBase0 = attn + (size_t)(s0 + li) * HID_;
  const _Float16* __restrict__ aBase1 = attn + (size_t)(s0 + 16 + li) * HID_;

  const unsigned lds0 = (unsigned)(uintptr_t)&wtile[0][0];
  const unsigned lds1 = (unsigned)(uintptr_t)&wtile[1][0];

  v8f acc[2][8] = {};
  stage_tile(Wbase, 0, lds0, tid);
  for (int kk = 0; kk < HID_; kk += 32) {
    const int buf = (kk >> 5) & 1;
    if (kk + 32 < HID_) {
      stage_tile(Wbase, kk + 32, buf ? lds0 : lds1, tid);
      asm volatile("s_wait_asynccnt 4" ::: "memory");
    } else {
      asm volatile("s_wait_asynccnt 0" ::: "memory");
    }
    __syncthreads();

    const int ka = kk + half * 8;
    v16h a0 = pack16(*(const v8h*)(aBase0 + ka), *(const v8h*)(aBase0 + ka + 16));
    v16h a1 = pack16(*(const v8h*)(aBase1 + ka), *(const v8h*)(aBase1 + ka + 16));
    const _Float16* Bl = &wtile[buf][0];
#pragma unroll
    for (int t = 0; t < 8; ++t) {
      const _Float16* bp = Bl + (t * 16 + li) * 32 + half * 8;
      v16h b = pack16(*(const v8h*)bp, *(const v8h*)(bp + 16));
      acc[0][t] = WMMA_F16(a0, b, acc[0][t]);
      acc[1][t] = WMMA_F16(a1, b, acc[1][t]);
    }
    __syncthreads();
  }
#pragma unroll
  for (int r = 0; r < 2; ++r)
#pragma unroll
    for (int t = 0; t < 8; ++t)
#pragma unroll
      for (int j = 0; j < 8; ++j)
        out[(size_t)(s0 + r * 16 + j + half * 8) * HID_ + g * D_ + t * 16 + li] =
            acc[r][t][j];
}

// ----------------------------------------------------------------------------------------
extern "C" void kernel_launch(void* const* d_in, const int* in_sizes, int n_in,
                              void* d_out, int out_size, void* d_ws, size_t ws_size,
                              hipStream_t stream) {
  const float* x    = (const float*)d_in[0];
  const float* imp  = (const float*)d_in[1];
  const float* cosT = (const float*)d_in[2];
  const float* sinT = (const float*)d_in[3];
  const float* Wq   = (const float*)d_in[4];
  const float* Wk   = (const float*)d_in[5];
  const float* Wv   = (const float*)d_in[6];
  const float* Wo   = (const float*)d_in[7];
  float* out = (float*)d_out;

  char* ws = (char*)d_ws;
  const size_t MAT = (size_t)NE_ * sizeof(_Float16);          // 8 MB
  _Float16* xh  = (_Float16*)(ws + 0 * MAT);
  _Float16* wqh = (_Float16*)(ws + 1 * MAT);
  _Float16* wkh = (_Float16*)(ws + 2 * MAT);
  _Float16* wvh = (_Float16*)(ws + 3 * MAT);
  _Float16* woh = (_Float16*)(ws + 4 * MAT);
  _Float16* Qh  = (_Float16*)(ws + 5 * MAT);
  _Float16* Kh  = (_Float16*)(ws + 6 * MAT);
  _Float16* Vt  = (_Float16*)(ws + 7 * MAT);
  _Float16* attn= (_Float16*)(ws + 8 * MAT);
  float*    spd = (float*)   (ws + 9 * MAT);

  prep_kernel<<<(NE_ + 255) / 256, 256, 0, stream>>>(x, imp, Wq, Wk, Wv, Wo,
                                                     xh, wqh, wkh, wvh, woh, spd);

  // 3 matrices x 16 row-blocks x 16 heads = 768 blocks of 128 threads
  qkv_rope_kernel<<<768, 128, 0, stream>>>(xh, wqh, wkh, wvh, cosT, sinT, Qh, Kh, Vt);

  // 128 qtiles x 16 heads = 2048 waves
  flash_kernel<<<2048 / 4, 128, 0, stream>>>(Qh, Kh, Vt, spd, attn);

  // 16 row-blocks x 16 col groups = 256 blocks
  out_proj_kernel<<<256, 128, 0, stream>>>(attn, woh, out);
}
